// FlexRoPEAttention_81913616269882
// MI455X (gfx1250) — matrix operand order repaired and checked by reading, not compiled, not run on hardware
//
#include <hip/hip_runtime.h>
#include <math.h>

// Fixed problem dims (from reference)
#define BB 4
#define TT 2048
#define CC 1024
#define HH 16
#define DD 64
#define NQKV 3072
#define MROWS (BB*TT)        // 8192

typedef __attribute__((ext_vector_type(16))) __bf16 v16bf;
typedef __attribute__((ext_vector_type(8)))  float  v8f;

union Frag16 { v16bf v; unsigned u[8]; };

// ---- CDNA5 async global->LDS copy (ASYNCcnt path, bypasses VGPRs) ----------
// VDST operand = 32-bit LDS byte offset (generic shared ptr truncated to 32b,
// ISA 10.2: LDS aperture address = addr[31:0]).
__device__ __forceinline__ void async_ld_b128(const void* lds_ptr, const void* gptr) {
  unsigned loff = (unsigned)(size_t)lds_ptr;
  unsigned long long ga = (unsigned long long)(size_t)gptr;
  asm volatile("global_load_async_to_lds_b128 %0, %1, off"
               :: "v"(loff), "v"(ga)
               : "memory");
}
__device__ __forceinline__ void wait_async0() {
  asm volatile("s_wait_asynccnt 0x0" ::: "memory");
}

// A-matrix 16x32 bf16 fragment from LDS, row-major K-contiguous storage.
// p0 = dword ptr at (row*strideDw + kDwOff + ((lane<16)?0:4))
// lanes<16 hold K 0..7 (dw 0..3) and K 16..23 (dw 8..11); lanes>=16 K 8..15 / 24..31.
__device__ __forceinline__ v16bf lds_afrag(const unsigned* p0) {
  Frag16 f;
#pragma unroll
  for (int i = 0; i < 4; ++i) f.u[i] = p0[i];
#pragma unroll
  for (int i = 0; i < 4; ++i) f.u[4 + i] = p0[8 + i];
  return f.v;
}

// B-matrix 32x16 bf16 fragment from LDS stored N-major with K contiguous.
// p0 = dword ptr at (ncol*strideDw + kDwOff + ((lane<16)?0:4))
// lanes<16 hold K 0..7 (dw 0..3) and K 16..23 (dw 8..11); lanes>=16 K 8..15 / 24..31.
__device__ __forceinline__ v16bf lds_bfrag(const unsigned* p0) {
  Frag16 f;
#pragma unroll
  for (int i = 0; i < 4; ++i) f.u[i] = p0[i];
#pragma unroll
  for (int i = 0; i < 4; ++i) f.u[4 + i] = p0[8 + i];
  return f.v;
}

// ---------------------------------------------------------------- convert
__global__ __launch_bounds__(256) void k_f32_to_bf16(const float* __restrict__ in,
                                                     unsigned short* __restrict__ outp,
                                                     int n) {
  __bf16* out = reinterpret_cast<__bf16*>(outp);
  int i = (blockIdx.x * 256 + threadIdx.x) * 4;
  if (i + 3 < n) {
    float4 v = *(const float4*)(in + i);
    out[i + 0] = (__bf16)v.x;
    out[i + 1] = (__bf16)v.y;
    out[i + 2] = (__bf16)v.z;
    out[i + 3] = (__bf16)v.w;
  }
}

// ---------------------------------------------------------------- GEMM (bf16 WMMA)
// C[M,N] = A[M,K] * B[K,N] + bias.  mode 0: scatter into q/k/v bf16 [B,H,T,D].
// mode 1: write fp32 to outf[M,CC].
__global__ __launch_bounds__(256) void k_gemm_bf16(const unsigned short* __restrict__ Ap,
                                                   const unsigned short* __restrict__ Bp,
                                                   const float* __restrict__ bias,
                                                   int K, int N, int mode,
                                                   unsigned short* __restrict__ qp,
                                                   unsigned short* __restrict__ kp,
                                                   unsigned short* __restrict__ vp,
                                                   float* __restrict__ outf) {
  const __bf16* A  = reinterpret_cast<const __bf16*>(Ap);
  const __bf16* Bw = reinterpret_cast<const __bf16*>(Bp);
  __bf16* qb = reinterpret_cast<__bf16*>(qp);
  __bf16* kb = reinterpret_cast<__bf16*>(kp);
  __bf16* vb = reinterpret_cast<__bf16*>(vp);

  __shared__ alignas(16) __bf16 As[64 * 32];    // row-major [64][32]
  __shared__ alignas(16) __bf16 Bst[128 * 32];  // N-major  [128][32] (K contiguous)

  const int tid  = threadIdx.x;
  const int lane = tid & 31;
  const int w    = tid >> 5;     // 8 waves
  const int wr   = w >> 2;       // 0..1  (rows, 32 each)
  const int wc   = w & 3;        // 0..3  (cols, 32 each)
  const int m0   = blockIdx.x * 64;
  const int n0   = blockIdx.y * 128;

  v8f acc[2][2] = {};

  const int arow = tid >> 2;          // 0..63
  const int acg  = (tid & 3) * 8;     // 0,8,16,24
  const int brow = tid >> 3;          // 0..31
  const int bcg  = (tid & 7) * 16;    // 0..112

  const int asel = (lane < 16) ? 0 : 4;
  const int bsel = (lane < 16) ? 0 : 4;
  const int l16  = lane & 15;

  for (int k0 = 0; k0 < K; k0 += 32) {
    // stage A tile (64x32): async global->LDS B128, one per thread (16 B)
    async_ld_b128(As + arow * 32 + acg,
                  A + (size_t)(m0 + arow) * K + k0 + acg);
    // stage B tile transposed: read 16 contiguous bf16, scatter as columns
    {
      union { uint4 q[2]; unsigned short s[16]; } tb;
      const __bf16* bs = Bw + (size_t)(k0 + brow) * N + n0 + bcg;
      tb.q[0] = *(const uint4*)(bs);
      tb.q[1] = *(const uint4*)(bs + 8);
      unsigned short* bst = (unsigned short*)Bst;
#pragma unroll
      for (int i = 0; i < 16; ++i) bst[(bcg + i) * 32 + brow] = tb.s[i];
    }
    if (k0 + 32 < K) {  // hint next K tiles (global_prefetch_b8)
      __builtin_prefetch(A + (size_t)(m0 + arow) * K + k0 + 32 + acg, 0, 1);
      __builtin_prefetch(Bw + (size_t)(k0 + 32 + brow) * N + n0 + bcg, 0, 1);
    }
    wait_async0();
    __syncthreads();

    const unsigned* Au = (const unsigned*)As;
    const unsigned* Bu = (const unsigned*)Bst;
    v16bf af[2], bfg[2];
#pragma unroll
    for (int mi = 0; mi < 2; ++mi)
      af[mi] = lds_afrag(Au + (wr * 32 + mi * 16 + l16) * 16 + asel);
#pragma unroll
    for (int ni = 0; ni < 2; ++ni)
      bfg[ni] = lds_bfrag(Bu + (wc * 32 + ni * 16 + l16) * 16 + bsel);
#pragma unroll
    for (int mi = 0; mi < 2; ++mi)
#pragma unroll
      for (int ni = 0; ni < 2; ++ni)
        acc[mi][ni] = __builtin_amdgcn_wmma_f32_16x16x32_bf16(
            false, af[mi], false, bfg[ni], (short)0, acc[mi][ni], false, false);
    __syncthreads();
  }

  // Epilogue.  C/D layout: N = lane%16, M = reg + (lane<16?0:8) within 16x16 tile.
  const int hi = (lane < 16) ? 0 : 8;
#pragma unroll
  for (int mi = 0; mi < 2; ++mi) {
#pragma unroll
    for (int ni = 0; ni < 2; ++ni) {
      int n = n0 + wc * 32 + ni * 16 + l16;
      float bv = bias[n];
#pragma unroll
      for (int r = 0; r < 8; ++r) {
        int m = m0 + wr * 32 + mi * 16 + r + hi;
        float val = acc[mi][ni][r] + bv;
        if (mode == 0) {
          int sel = n >> 10, rem = n & 1023;
          int h = rem >> 6, d = rem & 63;
          int b = m >> 11, t = m & 2047;
          size_t dst = (((size_t)(b * HH + h)) * TT + t) * DD + d;
          __bf16 o = (__bf16)val;
          if (sel == 0) qb[dst] = o;
          else if (sel == 1) kb[dst] = o;
          else vb[dst] = o;
        } else {
          outf[(size_t)m * CC + n] = val;
        }
      }
    }
  }
}

// ---------------------------------------------------------------- RoPE
__global__ __launch_bounds__(256) void k_rope(unsigned short* __restrict__ qp,
                                              unsigned short* __restrict__ kp,
                                              const int* __restrict__ keep,
                                              const int* __restrict__ olen) {
  __bf16* qb = reinterpret_cast<__bf16*>(qp);
  __bf16* kb = reinterpret_cast<__bf16*>(kp);
  int g = blockIdx.x * 256 + threadIdx.x;   // BB*HH*TT*32 threads
  int j = g & 31;
  int t = (g >> 5) & (TT - 1);
  int h = (g >> 16) & (HH - 1);
  int b = g >> 20;
  float pos  = (float)keep[b * TT + t];
  float invL = 1.0f / (float)(*olen);
  float frac = (float)j * (1.0f / 31.0f);
  float freq = 0.5f * powf(80.0f, frac);    // MIN*(MAX/MIN)^(j/(half-1))
  float ang  = 6.28318530717958647f * pos * invL * freq;
  float s, c;
  sincosf(ang, &s, &c);
  size_t base = (((size_t)(b * HH + h)) * TT + t) * DD + j;
  float q1 = (float)qb[base], q2 = (float)qb[base + 32];
  qb[base]      = (__bf16)(q1 * c - q2 * s);
  qb[base + 32] = (__bf16)(q1 * s + q2 * c);
  float k1 = (float)kb[base], k2 = (float)kb[base + 32];
  kb[base]      = (__bf16)(k1 * c - k2 * s);
  kb[base + 32] = (__bf16)(k1 * s + k2 * c);
}

// ---------------------------------------------------------------- flash attention
// One block per (b, h, 64-row q tile).  8 waves: wave w -> row subtile mt=w&3,
// column subtile pair dp=w>>2 (both for S (cols=keys) and O (cols=head dim)).
__global__ __launch_bounds__(256) void k_attn(const unsigned short* __restrict__ qp,
                                              const unsigned short* __restrict__ kp,
                                              const unsigned short* __restrict__ vp,
                                              unsigned short* __restrict__ aop) {
  const __bf16* qb = reinterpret_cast<const __bf16*>(qp);
  const __bf16* kb = reinterpret_cast<const __bf16*>(kp);
  const __bf16* vb = reinterpret_cast<const __bf16*>(vp);
  __bf16* ao = reinterpret_cast<__bf16*>(aop);

  __shared__ alignas(16) __bf16 Qs[64 * 64];  // [q][d] row-major
  __shared__ alignas(16) __bf16 Ks[64 * 64];  // [k][d] row-major (B frags for Q.K^T)
  __shared__ alignas(16) __bf16 Vt[64 * 64];  // [d][k] (B frags for P.V)
  __shared__ alignas(16) float  Sf[64 * 64];  // scores f32
  __shared__ alignas(16) __bf16 Ps[64 * 64];  // exp probs bf16, [q][k]
  __shared__ float sm[64], sl[64], sal[64];

  const int tid  = threadIdx.x;
  const int lane = tid & 31;
  const int w    = tid >> 5;
  const int mt   = w & 3;      // row subtile
  const int dp   = w >> 2;     // column subtile pair
  const int l16  = lane & 15;
  const int hi   = (lane < 16) ? 0 : 8;
  const int asel = (lane < 16) ? 0 : 4;
  const int bsel = (lane < 16) ? 0 : 4;

  const int qt = blockIdx.x & 31;
  const int h  = (blockIdx.x >> 5) & 15;
  const int b  = blockIdx.x >> 9;
  const size_t headBase = ((size_t)(b * HH + h)) * TT * DD;

  const int lrow = tid >> 2;           // 0..63
  const int lcg  = (tid & 3) * 16;     // 0,16,32,48

  {  // load Q tile: two async B128 per thread (32 B)
    const __bf16* src = qb + headBase + (size_t)(qt * 64 + lrow) * DD + lcg;
    async_ld_b128(Qs + lrow * 64 + lcg,     src);
    async_ld_b128(Qs + lrow * 64 + lcg + 8, src + 8);
  }
  if (tid < 64) { sm[tid] = -1e30f; sl[tid] = 0.0f; }
  v8f oacc[2] = {};
  wait_async0();
  __syncthreads();

  for (int kt = 0; kt < TT / 64; ++kt) {
    {  // load K row-major (async), V transposed (manual)
      const __bf16* ks = kb + headBase + (size_t)(kt * 64 + lrow) * DD + lcg;
      async_ld_b128(Ks + lrow * 64 + lcg,     ks);
      async_ld_b128(Ks + lrow * 64 + lcg + 8, ks + 8);
      union { uint4 q[2]; unsigned short s[16]; } tv;
      const uint4* vs = (const uint4*)(vb + headBase + (size_t)(kt * 64 + lrow) * DD + lcg);
      tv.q[0] = vs[0];
      tv.q[1] = vs[1];
      unsigned short* vt = (unsigned short*)Vt;
#pragma unroll
      for (int i = 0; i < 16; ++i) vt[(lcg + i) * 64 + lrow] = tv.s[i];
    }
    wait_async0();
    __syncthreads();

    // S = Q . K^T   (two 16x16 tiles per wave, K-dim = d, 2 steps of 32)
    {
      const unsigned* Qu = (const unsigned*)Qs;
      const unsigned* Ku = (const unsigned*)Ks;
      v8f sacc[2] = {};
#pragma unroll
      for (int ks = 0; ks < 2; ++ks) {
        v16bf aq = lds_afrag(Qu + (mt * 16 + l16) * 32 + ks * 16 + asel);
#pragma unroll
        for (int ni = 0; ni < 2; ++ni) {
          v16bf bk = lds_bfrag(Ku + ((dp * 2 + ni) * 16 + l16) * 32 + ks * 16 + bsel);
          sacc[ni] = __builtin_amdgcn_wmma_f32_16x16x32_bf16(
              false, aq, false, bk, (short)0, sacc[ni], false, false);
        }
      }
#pragma unroll
      for (int ni = 0; ni < 2; ++ni)
#pragma unroll
        for (int r = 0; r < 8; ++r)
          Sf[(mt * 16 + r + hi) * 64 + (dp * 2 + ni) * 16 + l16] = sacc[ni][r];
    }
    __syncthreads();

    // online softmax: row = tid/4, 4 lanes per row (width-4 shuffle reduce)
    {
      const int row = tid >> 2, sub = tid & 3;
      const float scale = 0.125f;  // 1/sqrt(64)
      float mold = sm[row];
      float sv[16];
      float tmax = -1e30f;
#pragma unroll
      for (int i = 0; i < 16; ++i) {
        sv[i] = Sf[row * 64 + sub * 16 + i] * scale;
        tmax = fmaxf(tmax, sv[i]);
      }
      tmax = fmaxf(tmax, __shfl_xor(tmax, 1, 4));
      tmax = fmaxf(tmax, __shfl_xor(tmax, 2, 4));
      float mnew = fmaxf(mold, tmax);
      float psum = 0.0f;
#pragma unroll
      for (int i = 0; i < 16; ++i) {
        float p = __expf(sv[i] - mnew);
        psum += p;
        Ps[row * 64 + sub * 16 + i] = (__bf16)p;
      }
      psum += __shfl_xor(psum, 1, 4);
      psum += __shfl_xor(psum, 2, 4);
      float alpha = __expf(mold - mnew);
      if (sub == 0) {
        sm[row] = mnew;
        sl[row] = sl[row] * alpha + psum;
        sal[row] = alpha;
      }
    }
    __syncthreads();

    // rescale O by alpha[row], then O += P . V
#pragma unroll
    for (int ni = 0; ni < 2; ++ni)
#pragma unroll
      for (int r = 0; r < 8; ++r)
        oacc[ni][r] *= sal[mt * 16 + r + hi];
    {
      const unsigned* Pu = (const unsigned*)Ps;
      const unsigned* Vu = (const unsigned*)Vt;
#pragma unroll
      for (int ks = 0; ks < 2; ++ks) {
        v16bf ap = lds_afrag(Pu + (mt * 16 + l16) * 32 + ks * 16 + asel);
#pragma unroll
        for (int ni = 0; ni < 2; ++ni) {
          v16bf bv = lds_bfrag(Vu + ((dp * 2 + ni) * 16 + l16) * 32 + ks * 16 + bsel);
          oacc[ni] = __builtin_amdgcn_wmma_f32_16x16x32_bf16(
              false, ap, false, bv, (short)0, oacc[ni], false, false);
        }
      }
    }
    __syncthreads();
  }

  // normalize and write to attn-out [B*T, H*D] bf16
#pragma unroll
  for (int ni = 0; ni < 2; ++ni)
#pragma unroll
    for (int r = 0; r < 8; ++r) {
      int row = mt * 16 + r + hi;
      float val = oacc[ni][r] / sl[row];
      int t = qt * 64 + row;
      int d = (dp * 2 + ni) * 16 + l16;
      ao[((size_t)(b * TT + t)) * CC + h * DD + d] = (__bf16)val;
    }
}

// ---------------------------------------------------------------- launch
extern "C" void kernel_launch(void* const* d_in, const int* in_sizes, int n_in,
                              void* d_out, int out_size, void* d_ws, size_t ws_size,
                              hipStream_t stream) {
  (void)in_sizes; (void)n_in; (void)out_size; (void)ws_size;
  const float* x     = (const float*)d_in[0];
  const int*   keep  = (const int*)d_in[1];
  const int*   olen  = (const int*)d_in[2];
  const float* Wqkv  = (const float*)d_in[3];
  const float* bqkv  = (const float*)d_in[4];
  const float* Wproj = (const float*)d_in[5];
  const float* bproj = (const float*)d_in[6];
  float* out = (float*)d_out;

  char* ws = (char*)d_ws;
  unsigned short* xb  = (unsigned short*)(ws);                    // 16 MB  x bf16
  unsigned short* wqb = (unsigned short*)(ws + 16777216);         //  6 MB  W_qkv bf16
  unsigned short* wpb = (unsigned short*)(ws + 23068672);         //  2 MB  W_proj bf16
  unsigned short* qb  = (unsigned short*)(ws + 25165824);         // 16 MB  q [B,H,T,D]
  unsigned short* kb  = (unsigned short*)(ws + 41943040);         // 16 MB  k
  unsigned short* vb  = (unsigned short*)(ws + 58720256);         // 16 MB  v
  unsigned short* ao  = (unsigned short*)(ws + 75497472);         // 16 MB  attn out

  k_f32_to_bf16<<<(MROWS * CC / 4) / 256, 256, 0, stream>>>(x, xb, MROWS * CC);
  k_f32_to_bf16<<<(CC * NQKV / 4) / 256, 256, 0, stream>>>(Wqkv, wqb, CC * NQKV);
  k_f32_to_bf16<<<(CC * CC / 4) / 256, 256, 0, stream>>>(Wproj, wpb, CC * CC);

  dim3 g1(MROWS / 64, NQKV / 128);
  k_gemm_bf16<<<g1, 256, 0, stream>>>(xb, wqb, bqkv, CC, NQKV, 0, qb, kb, vb, nullptr);

  k_rope<<<(BB * HH * TT * 32) / 256, 256, 0, stream>>>(qb, kb, keep, olen);

  k_attn<<<BB * HH * (TT / 64), 256, 0, stream>>>(qb, kb, vb, ao);

  dim3 g2(MROWS / 64, CC / 128);
  k_gemm_bf16<<<g2, 256, 0, stream>>>(ao, wpb, bproj, CC, CC, 1, nullptr, nullptr, nullptr, out);
}
